// EcgServer_85684597555564
// MI455X (gfx1250) — matrix operand order repaired and checked by reading, not compiled, not run on hardware
//
#include <hip/hip_runtime.h>

// ---------------------------------------------------------------------------
// MI455X (gfx1250) fused MLP:  out = poly(x @ W3^T + b3) @ W4^T + b4
//   B=131072, D_IN=512, D_HID=256, D_OUT=5
// Streaming, memory-bound design: enc_x read once (268 MB -> ~12us @ 23.3TB/s),
// W3 staged in LDS as bf16 (260 KB of 320 KB WGP LDS), GEMM1 via
// v_wmma_f32_16x16x32_bf16, bias/poly/GEMM2 fused in registers.
// This revision software-pipelines both the LDS B-fragment loads (double
// buffer, load t+2 before wmma t) and the global A loads (k0+32 issued before
// k0 is consumed) so ds/vmem latency hides behind WMMA issue.
// ---------------------------------------------------------------------------

typedef __attribute__((ext_vector_type(16))) __bf16 v16bf;
typedef __attribute__((ext_vector_type(8)))  __bf16 v8bf;
typedef __attribute__((ext_vector_type(8)))  float  v8f;

#define D_IN   512
#define D_HID  256
#define D_OUT  5
#define ROWS_PER_BLOCK 128                 // 8 waves * 16 rows
#define W3_STRIDE 520                      // 512 + 8 bf16 pad: 1040B row -> 4-bank shift, conflict-free
#define LDS_W3_BYTES (D_HID * W3_STRIDE * 2)   // 266240

__global__ __launch_bounds__(256)
void EcgServer_85684597555564_kernel(const float* __restrict__ enc_x,
                                     const float* __restrict__ W3,
                                     const float* __restrict__ b3,
                                     const float* __restrict__ W4,
                                     const float* __restrict__ b4,
                                     float* __restrict__ out) {
    extern __shared__ char smem[];
    __bf16* lds_w3 = (__bf16*)smem;                       // [256][520] bf16
    float*  lds_b3 = (float*)(smem + LDS_W3_BYTES);       // [256]
    float*  lds_w4 = lds_b3 + D_HID;                      // [5][256]
    float*  lds_b4 = lds_w4 + D_OUT * D_HID;              // [5]

    const int tid = threadIdx.x;

    // ---- Stage W3 (fp32 -> bf16, padded rows) + consts into LDS ----------
    #pragma unroll 4
    for (int i = tid; i < (D_HID * D_IN) / 4; i += 256) {
        const int e   = i * 4;
        const int row = e >> 9;          // / 512
        const int col = e & 511;
        const float4 v = *(const float4*)(W3 + e);
        __bf16* p = lds_w3 + row * W3_STRIDE + col;
        p[0] = (__bf16)v.x; p[1] = (__bf16)v.y;
        p[2] = (__bf16)v.z; p[3] = (__bf16)v.w;
    }
    if (tid < D_HID) lds_b3[tid] = b3[tid];
    for (int i = tid; i < D_OUT * D_HID; i += 256) lds_w4[i] = W4[i];
    if (tid < D_OUT) lds_b4[tid] = b4[tid];
    __syncthreads();

    // ---- Per-wave tile: 16 B-rows x 256 hidden -----------------------------
    const int wave = tid >> 5;
    const int lane = tid & 31;
    const int half = lane >> 4;          // 0: lanes 0-15, 1: lanes 16-31
    const int l16  = lane & 15;
    const int m0   = blockIdx.x * ROWS_PER_BLOCK + wave * 16;

    // A-fragment source row for this lane (ISA 16-bit A 16x32 layout:
    // lanes 0-15 hold K {k0..k0+7, k0+16..k0+23}; lanes 16-31 the +8 halves).
    const float* arow = enc_x + (size_t)(m0 + l16) * D_IN;

    // B fragment fetch (32x16 tile): lane holds column N = t*16 + l16,
    // K = k0 + half*8 .. +7 and k0 + 16 + half*8 .. +7 of W3 row N in LDS.
    const __bf16* bcol = lds_w3 + (size_t)l16 * W3_STRIDE + half * 8;
    auto loadB = [&](int t, int k0) -> v16bf {
        const __bf16* bp = bcol + t * (16 * W3_STRIDE) + k0;
        const v8bf blo = *(const v8bf*)bp;
        const v8bf bhi = *(const v8bf*)(bp + 16);
        return __builtin_shufflevector(blo, bhi,
            0,1,2,3,4,5,6,7,8,9,10,11,12,13,14,15);
    };

    v8f acc[16];
    const v8f vzero = {0.f, 0.f, 0.f, 0.f, 0.f, 0.f, 0.f, 0.f};
    #pragma unroll
    for (int t = 0; t < 16; ++t) acc[t] = vzero;

    // Prime the A pipeline (k0 = 0).
    const float* ap0 = arow + half * 8;
    float4 ca0 = *(const float4*)(ap0);
    float4 ca1 = *(const float4*)(ap0 + 4);
    float4 ca2 = *(const float4*)(ap0 + 16);
    float4 ca3 = *(const float4*)(ap0 + 20);

    #pragma unroll 2
    for (int k0 = 0; k0 < D_IN; k0 += 32) {
        // Issue next k-step's global A loads before consuming current data.
        float4 na0 = ca0, na1 = ca1, na2 = ca2, na3 = ca3;
        if (k0 + 32 < D_IN) {
            const float* np = arow + (k0 + 32) + half * 8;
            na0 = *(const float4*)(np);
            na1 = *(const float4*)(np + 4);
            na2 = *(const float4*)(np + 16);
            na3 = *(const float4*)(np + 20);
        }
        if (k0 + 64 < D_IN)
            __builtin_prefetch(arow + k0 + 64, 0, 0);     // global_prefetch_b8

        // Convert current A data to the bf16 fragment.
        v16bf a;
        a[0]=(__bf16)ca0.x;  a[1]=(__bf16)ca0.y;  a[2]=(__bf16)ca0.z;  a[3]=(__bf16)ca0.w;
        a[4]=(__bf16)ca1.x;  a[5]=(__bf16)ca1.y;  a[6]=(__bf16)ca1.z;  a[7]=(__bf16)ca1.w;
        a[8]=(__bf16)ca2.x;  a[9]=(__bf16)ca2.y;  a[10]=(__bf16)ca2.z; a[11]=(__bf16)ca2.w;
        a[12]=(__bf16)ca3.x; a[13]=(__bf16)ca3.y; a[14]=(__bf16)ca3.z; a[15]=(__bf16)ca3.w;

        // B-fragment double buffer: load t+2 before issuing wmma t, so the
        // LDS latency hides behind one WMMA instead of stalling every tile.
        v16bf bf0 = loadB(0, k0);
        v16bf bf1 = loadB(1, k0);
        #pragma unroll
        for (int t = 0; t < 16; ++t) {
            const v16bf bcur = (t & 1) ? bf1 : bf0;
            if (t + 2 < 16) {
                if (t & 1) bf1 = loadB(t + 2, k0);
                else       bf0 = loadB(t + 2, k0);
            }
            acc[t] = __builtin_amdgcn_wmma_f32_16x16x32_bf16(
                false, a, false, bcur, (short)0, acc[t], false, false);
        }

        ca0 = na0; ca1 = na1; ca2 = na2; ca3 = na3;
    }

    // ---- Fused epilogue: bias + poly + (256 -> 5) GEMM in registers --------
    // C layout: VGPR j holds M=j (lanes 0-15) / M=j+8 (lanes 16-31), N=l16.
    float po[8][D_OUT];
    #pragma unroll
    for (int j = 0; j < 8; ++j)
        #pragma unroll
        for (int o = 0; o < D_OUT; ++o) po[j][o] = 0.f;

    #pragma unroll
    for (int t = 0; t < 16; ++t) {
        const int c = t * 16 + l16;
        const float bias = lds_b3[c];
        float w4v[D_OUT];
        #pragma unroll
        for (int o = 0; o < D_OUT; ++o) w4v[o] = lds_w4[o * D_HID + c];
        #pragma unroll
        for (int j = 0; j < 8; ++j) {
            float h = acc[t][j] + bias;
            h = 0.854f + h * (0.505f + 0.0654f * h);   // approx leaky relu
            #pragma unroll
            for (int o = 0; o < D_OUT; ++o) po[j][o] += h * w4v[o];
        }
    }

    // Reduce over the 16 lanes of each half (butterfly stays inside halves),
    // add b4, store: lane l16==o writes out[row][o].
    #pragma unroll
    for (int j = 0; j < 8; ++j) {
        const int row = m0 + j + half * 8;
        float* orow = out + (size_t)row * D_OUT;
        #pragma unroll
        for (int o = 0; o < D_OUT; ++o) {
            float v = po[j][o];
            v += __shfl_xor(v, 1, 32);
            v += __shfl_xor(v, 2, 32);
            v += __shfl_xor(v, 4, 32);
            v += __shfl_xor(v, 8, 32);
            v += lds_b4[o];
            if (l16 == o) orow[o] = v;
        }
    }
}

extern "C" void kernel_launch(void* const* d_in, const int* in_sizes, int n_in,
                              void* d_out, int out_size, void* d_ws, size_t ws_size,
                              hipStream_t stream) {
    (void)in_sizes; (void)n_in; (void)d_ws; (void)ws_size; (void)out_size;
    const float* enc_x = (const float*)d_in[0];
    const float* W3    = (const float*)d_in[1];
    const float* b3    = (const float*)d_in[2];
    const float* W4    = (const float*)d_in[3];
    const float* b4    = (const float*)d_in[4];
    float* out = (float*)d_out;

    const int B = 131072;
    const dim3 grid(B / ROWS_PER_BLOCK);          // 1024 workgroups
    const dim3 block(256);                        // 8 waves (wave32)
    const size_t shmem = LDS_W3_BYTES + (D_HID + D_OUT * D_HID + D_OUT) * sizeof(float);
    EcgServer_85684597555564_kernel<<<grid, block, shmem, stream>>>(
        enc_x, W3, b3, W4, b4, out);
}
